// DeepHedgingModel_24086176596631
// MI455X (gfx1250) — matrix operand order repaired and checked by reading, not compiled, not run on hardware
//
#include <hip/hip_runtime.h>
#include <hip/hip_fp16.h>

typedef __attribute__((ext_vector_type(16))) _Float16 v16h;
typedef __attribute__((ext_vector_type(8)))  _Float16 v8h;
typedef __attribute__((ext_vector_type(8)))  float    v8f;
typedef __attribute__((ext_vector_type(2)))  float    v2f;

#define T_STEPS 512
#define FEAT    4
#define INF     5     // F + 1 (prev delta channel)
#define HID     32

// ---- activations on raw HW transcendentals ----
#if __has_builtin(__builtin_amdgcn_tanhf)
__device__ __forceinline__ float tanh_f(float x) {
    return __builtin_amdgcn_tanhf(x);            // single v_tanh_f32
}
__device__ __forceinline__ float sigm_f(float x) {
    return fmaf(0.5f, __builtin_amdgcn_tanhf(0.5f * x), 0.5f);
}
#else
__device__ __forceinline__ float sigm_f(float x) {
    return __builtin_amdgcn_rcpf(1.0f + __expf(-x));
}
__device__ __forceinline__ float tanh_f(float x) {
    return 1.0f - 2.0f * __builtin_amdgcn_rcpf(__expf(2.0f * x) + 1.0f);
}
#endif

__device__ __forceinline__ v8f wmma_f16(v16h a, v16h b, v8f c) {
    return __builtin_amdgcn_wmma_f32_16x16x32_f16(
        false, a, false, b, (short)0, c, false, false);
}
__device__ __forceinline__ v8f wmma_f32k4(v2f a, v2f b, v8f c) {
    return __builtin_amdgcn_wmma_f32_16x16x4_f32(
        false, a, false, b, (short)0, c, false, false);
}

// Two wave32s cooperate on one 16-row batch tile: wave "half" h owns hidden
// units 16h..16h+15 (gate tiles i,f,g,o = q*2+h) and its half of the cell
// state. Halves meet only at the hidden vector: each publishes 16 columns of
// h into a double-buffered LDS tile, one __syncthreads per step, then both
// rebuild the full f16 A-fragment. The tiny decision MLP is recomputed by
// both waves (cheaper than a second barrier). 1024 waves total.
__global__ __launch_bounds__(256)
void deep_hedge_lstm_kernel(const float* __restrict__ x,
                            const float* __restrict__ bn_gamma,
                            const float* __restrict__ bn_beta,
                            const float* __restrict__ bn_mean,
                            const float* __restrict__ bn_var,
                            const float* __restrict__ W_ih,   // [128,5]
                            const float* __restrict__ b_ih,   // [128]
                            const float* __restrict__ W_hh,   // [128,32]
                            const float* __restrict__ b_hh,   // [128]
                            const float* __restrict__ W1,     // [32,32]
                            const float* __restrict__ b1,     // [32]
                            const float* __restrict__ W2,     // [1,32]
                            const float* __restrict__ b2,     // [1]
                            float* __restrict__ out,          // [B,T,1]
                            int B)
{
    const int lane   = threadIdx.x & 31;
    const int wave   = threadIdx.x >> 5;     // 0..7
    const int pairId = wave >> 1;            // row-tile within block, 0..3
    const int half   = wave & 1;             // hidden half owned by this wave
    const int rowBase = (blockIdx.x * 4 + pairId) * 16;

    // double-buffered h tiles (one barrier per step), per-wave scalar bufs
    __shared__ _Float16 hbuf[2][4][16][32];
    __shared__ float    dbuf[8][16];         // raw MLP layer-2 row sums
    __shared__ float    zdbuf[8][16];        // batchnormed prev-delta per row

    const int ncol   = lane & 15;            // fragment column / row id
    const bool loHalf = (lane < 16);
    const int kbase = loHalf ? 0 : 16;       // f16 B-frag K base
    const int kb2   = loHalf ? 0 : 2;        // f32 K4-frag K base
    const int mbase = loHalf ? 0 : 8;        // D-layout row base
    const int kb_a  = loHalf ? 0 : 16;       // f16 A-frag K base

    // ---- this wave's 4 gate tiles (i,f,g,o of its hidden half) ----
    v16h Bhh[4];       // W_hh.T tiles (f16, K=32)
    v2f  Bihf[4];      // W_ih.T feature tiles (f32, K=4)
    float wih4[4];     // W_ih[:,4] (prev-delta column)
    float biasG[4];
    #pragma unroll
    for (int q = 0; q < 4; ++q) {
        const int col = (q * 2 + half) * 16 + ncol;   // gate output 0..127
        v16h bh;
        #pragma unroll
        for (int j = 0; j < 16; ++j)
            bh[j] = (_Float16)W_hh[col * HID + kbase + j];
        Bhh[q] = bh;
        v2f bf;
        bf[0] = W_ih[col * INF + kb2];
        bf[1] = W_ih[col * INF + kb2 + 1];
        Bihf[q] = bf;
        wih4[q] = W_ih[col * INF + 4];
        biasG[q] = b_ih[col] + b_hh[col];
    }
    // ---- decision MLP weights (both waves keep the full MLP) ----
    v16h Bw1[2];
    float biasM[2], w2c[2];
    #pragma unroll
    for (int tN = 0; tN < 2; ++tN) {
        const int col = tN * 16 + ncol;               // MLP hidden 0..31
        v16h bw;
        #pragma unroll
        for (int j = 0; j < 16; ++j)
            bw[j] = (_Float16)W1[col * HID + kbase + j];
        Bw1[tN] = bw;
        biasM[tN] = b1[col];
        w2c[tN]  = W2[col];
    }

    // batchnorm folded to affine: z = raw*sc + sh
    float sc[INF], sh[INF];
    #pragma unroll
    for (int k = 0; k < INF; ++k) {
        const float inv = rsqrtf(bn_var[k] + 1e-5f);
        sc[k] = inv * bn_gamma[k];
        sh[k] = bn_beta[k] - bn_mean[k] * sc[k];
    }
    const float b2s = b2[0];

    // ---- recurrent state ----
    v8f  cst = {};           // this half's cell state, D layout
    v16h Ah  = {};           // full h as f16 A-fragment, starts at 0
    if (loHalf) zdbuf[wave][ncol] = sh[4];   // d=0 at t=0 -> bn value sh[4]

    // both waves of the pair load the same x row (L0-cached duplicate)
    const float* xrow = x   + (size_t)(rowBase + ncol) * T_STEPS * FEAT;
    float*       orow = out + (size_t)(rowBase + ncol) * T_STEPS;

    for (int t = 0; t < T_STEPS; ++t) {
        // ---- batchnormed prev-delta for this lane's 8 rows (own copy) ----
        float zd[8];
        #pragma unroll
        for (int r = 0; r < 8; ++r) zd[r] = zdbuf[wave][mbase + r];

        // ---- batchnormed market features, f32 K4 A-fragment ----
        const float4 xv = *(const float4*)(xrow + (size_t)t * FEAT);
        if (t + 8 < T_STEPS)
            __builtin_prefetch(xrow + (size_t)(t + 8) * FEAT, 0, 3);
        v2f Azf;
        Azf[0] = (loHalf ? xv.x : xv.z) * sc[kb2]     + sh[kb2];
        Azf[1] = (loHalf ? xv.y : xv.w) * sc[kb2 + 1] + sh[kb2 + 1];

        // ---- 4 gate tiles of this hidden half ----
        v8f gi, gf, gg, go;
        #pragma unroll
        for (int q = 0; q < 4; ++q) {
            v8f C;
            #pragma unroll
            for (int r = 0; r < 8; ++r)
                C[r] = fmaf(zd[r], wih4[q], biasG[q]);   // bias + delta chan.
            C = wmma_f32k4(Azf, Bihf[q], C);             // + x @ W_ihx.T
            C = wmma_f16(Ah, Bhh[q], C);                 // + h @ W_hh.T
            if      (q == 0) gi = C;
            else if (q == 1) gf = C;
            else if (q == 2) gg = C;
            else             go = C;
        }

        // ---- LSTM pointwise update for this half ----
        v8f hv;
        #pragma unroll
        for (int r = 0; r < 8; ++r) {
            const float cn = sigm_f(gf[r]) * cst[r]
                           + sigm_f(gi[r]) * tanh_f(gg[r]);
            cst[r] = cn;
            hv[r] = sigm_f(go[r]) * tanh_f(cn);
        }

        // ---- publish own hidden half (columns half*16 .. half*16+15) ----
        _Float16* htile = &hbuf[t & 1][pairId][0][0];
        #pragma unroll
        for (int r = 0; r < 8; ++r)
            htile[(mbase + r) * 32 + half * 16 + ncol] = (_Float16)hv[r];

        __syncthreads();   // pair exchange (double buffer -> one barrier/step)

        // ---- full h -> f16 A-fragment ----
        {
            const _Float16* hrow = htile + ncol * 32;
            const v8h lo = *(const v8h*)(hrow + kb_a);        // K=kb..kb+7
            const v8h hi = *(const v8h*)(hrow + kb_a + 8);    // K=kb+8..+15
            Ah = __builtin_shufflevector(lo, hi, 0, 1, 2, 3, 4, 5, 6, 7,
                                                 8, 9, 10, 11, 12, 13, 14, 15);
        }

        // ---- decision MLP (redundant in both waves; avoids 2nd barrier) --
        v8f Cm0, Cm1;
        #pragma unroll
        for (int r = 0; r < 8; ++r) { Cm0[r] = biasM[0]; Cm1[r] = biasM[1]; }
        Cm0 = wmma_f16(Ah, Bw1[0], Cm0);
        Cm1 = wmma_f16(Ah, Bw1[1], Cm1);

        float rs[8];
        #pragma unroll
        for (int r = 0; r < 8; ++r) {
            float v = fmaxf(Cm0[r], 0.0f) * w2c[0]
                    + fmaxf(Cm1[r], 0.0f) * w2c[1];
            v += __shfl_xor(v, 1, 32);
            v += __shfl_xor(v, 2, 32);
            v += __shfl_xor(v, 4, 32);
            v += __shfl_xor(v, 8, 32);   // stays within each 16-lane half
            rs[r] = v;
        }
        if (ncol == 0) {                 // lane 0 -> rows 0-7, lane 16 -> 8-15
            #pragma unroll
            for (int r = 0; r < 8; ++r) dbuf[wave][mbase + r] = rs[r];
        }
        if (loHalf) {
            const float d = sigm_f(dbuf[wave][ncol] + b2s);
            if (half == 0) orow[t] = d;                 // single writer
            zdbuf[wave][ncol] = fmaf(d, sc[4], sh[4]);  // feeds next step
        }
    }
}

extern "C" void kernel_launch(void* const* d_in, const int* in_sizes, int n_in,
                              void* d_out, int out_size, void* d_ws, size_t ws_size,
                              hipStream_t stream) {
    const float* x        = (const float*)d_in[0];
    const float* bn_gamma = (const float*)d_in[1];
    const float* bn_beta  = (const float*)d_in[2];
    const float* bn_mean  = (const float*)d_in[3];
    const float* bn_var   = (const float*)d_in[4];
    const float* W_ih     = (const float*)d_in[5];
    const float* b_ih     = (const float*)d_in[6];
    const float* W_hh     = (const float*)d_in[7];
    const float* b_hh     = (const float*)d_in[8];
    const float* W1       = (const float*)d_in[9];
    const float* b1       = (const float*)d_in[10];
    const float* W2       = (const float*)d_in[11];
    const float* b2       = (const float*)d_in[12];
    float* out = (float*)d_out;

    const int B = in_sizes[0] / (T_STEPS * FEAT);   // 8192
    const int rowsPerBlock = 4 * 16;                // 4 row-tiles, 2 waves each
    const int nBlocks = (B + rowsPerBlock - 1) / rowsPerBlock;   // 128

    deep_hedge_lstm_kernel<<<nBlocks, 256, 0, stream>>>(
        x, bn_gamma, bn_beta, bn_mean, bn_var,
        W_ih, b_ih, W_hh, b_hh, W1, b1, W2, b2, out, B);
}
